// Transformer_66649302499497
// MI455X (gfx1250) — matrix operand order repaired and checked by reading, not compiled, not run on hardware
//
#include <hip/hip_runtime.h>
#include <cstdint>
#include <cstddef>

// ---------------- constants ----------------
#define DIMD      1024
#define NH        16
#define QKD       96
#define NOPE_D    64
#define ROPE_D    32
#define VD        64
#define KVL       256
#define INTER_D   4096
#define MOE_I     512
#define NEXP      16
#define OUTD      512
#define BATCH     4
#define SEQ       1024
#define TTOK      4096          // BATCH*SEQ
#define ROPE_BASE 10000.0f

typedef __attribute__((ext_vector_type(16))) __bf16 v16bf;
typedef __attribute__((ext_vector_type(8)))  float  v8f;

// ---------------- device helpers ----------------
static __device__ inline unsigned short f2bf(float f) {
    union { float f; unsigned u; } v; v.f = f;
    unsigned r = v.u + 0x7FFFu + ((v.u >> 16) & 1u);
    return (unsigned short)(r >> 16);
}

static __device__ inline v8f zero8() {
    v8f z = {0.f,0.f,0.f,0.f,0.f,0.f,0.f,0.f};
    return z;
}

static __device__ inline void lds_fence() {
    asm volatile("s_wait_dscnt 0" ::: "memory");
}

// A-fragment (16x32 bf16, M x K) direct from global. Per ISA 7.12.2: lane M=lane&15;
// halves 0..7 -> K = base+0..7, halves 8..15 -> K = base+16..23, base = (lane>=16)?8:0
static __device__ inline v16bf load_a_frag(const unsigned short* A, int row, int K,
                                           int k0, int lane) {
    const unsigned short* p = A + (size_t)row * K + k0 + ((lane >= 16) ? 8 : 0);
    union { v16bf v; uint4 q[2]; } f;
    f.q[0] = *reinterpret_cast<const uint4*>(p);
    f.q[1] = *reinterpret_cast<const uint4*>(p + 16);
    return f.v;
}

// B-fragment (32x16 bf16, K x N) where B[k][n] = W[n][k], W row-major (N,K).
// halves 0..7 -> K = base+0..7, halves 8..15 -> K = base+16..23, base = (lane>=16)?8:0 ; N = n (lane&15)
static __device__ inline v16bf load_b_frag(const unsigned short* W, int n, int K,
                                           int k0, int lane) {
    const unsigned short* p = W + (size_t)n * K + k0 + ((lane >= 16) ? 8 : 0);
    union { v16bf v; uint4 q[2]; } f;
    f.q[0] = *reinterpret_cast<const uint4*>(p);
    f.q[1] = *reinterpret_cast<const uint4*>(p + 16);
    return f.v;
}

// ---------------- elementwise / norm kernels ----------------
__global__ __launch_bounds__(256) void f32_to_bf16_k(const float* s, unsigned short* d, size_t n) {
    size_t i = (size_t)blockIdx.x * blockDim.x + threadIdx.x;
    if (i < n) d[i] = f2bf(s[i]);
}

__global__ __launch_bounds__(256) void zero_f32_k(float* p, size_t n) {
    size_t i = (size_t)blockIdx.x * blockDim.x + threadIdx.x;
    if (i < n) p[i] = 0.f;
}

__global__ __launch_bounds__(256) void add_f32_k(float* a, const float* b, size_t n) {
    size_t i = (size_t)blockIdx.x * blockDim.x + threadIdx.x;
    if (i < n) a[i] += b[i];
}

// rmsnorm over D elements, input row stride `stride`; bf16 and optional f32 outputs (packed D)
__global__ __launch_bounds__(256) void rmsnorm_k(const float* x, const float* w, int D, int stride,
                                                 unsigned short* outB, float* outF) {
    int t = blockIdx.x;
    const float* row = x + (size_t)t * stride;
    __shared__ float red[256];
    float s = 0.f;
    for (int i = threadIdx.x; i < D; i += 256) { float v = row[i]; s += v * v; }
    red[threadIdx.x] = s; __syncthreads();
    for (int o = 128; o > 0; o >>= 1) {
        if (threadIdx.x < o) red[threadIdx.x] += red[threadIdx.x + o];
        __syncthreads();
    }
    float r = rsqrtf(red[0] / (float)D + 1e-6f);
    for (int i = threadIdx.x; i < D; i += 256) {
        float v = row[i] * r * w[i];
        if (outB) outB[(size_t)t * D + i] = f2bf(v);
        if (outF) outF[(size_t)t * D + i] = v;
    }
}

// ---------------- WMMA GEMM, LDS double-buffered, software-pipelined ----------------
// C(M,N) = A(M,K) @ W(N,K)^T
// mode 0: outF = acc + (addC ? addC : 0)
// mode 2: outBF = bf16( silu(gateSrc) * acc )
// Block: 256 thr = 8 waves (4 in M x 2 in N), tile 64M x 128N, K-step 64 (2 WMMA sub-slices).
// Pipeline: global->regs (top of iter, issued as a clause) overlaps WMMAs; regs->LDS at bottom.
// Prefetch values live in a named-member struct (reliable SROA) and __launch_bounds__(256,1)
// gives the allocator the full VGPR file so nothing spills to scratch.
#define KSTR 72   // padded LDS row stride (halfwords): 144B, 16B-aligned, bank stride 36
struct TileRegs { uint4 a0, a1, b0, b1, b2, b3; };

__global__ __launch_bounds__(256, 1) void gemm_bf16_k(const unsigned short* __restrict__ A,
                                                      const unsigned short* __restrict__ W,
                                                      int M, int N, int K,
                                                      const float* addC, const float* gateSrc,
                                                      float* outF, unsigned short* outBF,
                                                      int mode) {
    __shared__ unsigned short As[2][64 * KSTR];
    __shared__ unsigned short Bs[2][128 * KSTR];
    int tid  = threadIdx.x;
    int wave = tid >> 5, lane = tid & 31;
    int wm = wave & 3, wn = wave >> 2;
    int mblock = blockIdx.y * 64;
    int nblock = blockIdx.x * 128;
    int mbase  = mblock + wm * 16;
    int nbase0 = nblock + wn * 64;

    // 16B chunk mapping for one K=64 slice: A 512 chunks (2/thr), B 1024 chunks (4/thr);
    // chunk c: row = c>>3, sub = c&7  (sub*8 halfwords = sub*16 bytes within the row slice)
    int ar0 = tid >> 3,          as0 = tid & 7;
    int ar1 = (tid + 256) >> 3,  as1 = (tid + 256) & 7;
    int ga0 = mblock + ar0; if (ga0 >= M) ga0 = M - 1;
    int ga1 = mblock + ar1; if (ga1 >= M) ga1 = M - 1;
    int br0 = ar0, bs0 = as0, br1 = ar1, bs1 = as1;
    int br2 = (tid + 512) >> 3,  bs2 = (tid + 512) & 7;
    int br3 = (tid + 768) >> 3,  bs3 = (tid + 768) & 7;
    int gb0 = nblock + br0; if (gb0 >= N) gb0 = N - 1;
    int gb1 = nblock + br1; if (gb1 >= N) gb1 = N - 1;
    int gb2 = nblock + br2; if (gb2 >= N) gb2 = N - 1;
    int gb3 = nblock + br3; if (gb3 >= N) gb3 = N - 1;

    auto load_tile = [&](int k0) -> TileRegs {
        TileRegs t;
        t.a0 = *reinterpret_cast<const uint4*>(A + (size_t)ga0 * K + k0 + as0 * 8);
        t.a1 = *reinterpret_cast<const uint4*>(A + (size_t)ga1 * K + k0 + as1 * 8);
        t.b0 = *reinterpret_cast<const uint4*>(W + (size_t)gb0 * K + k0 + bs0 * 8);
        t.b1 = *reinterpret_cast<const uint4*>(W + (size_t)gb1 * K + k0 + bs1 * 8);
        t.b2 = *reinterpret_cast<const uint4*>(W + (size_t)gb2 * K + k0 + bs2 * 8);
        t.b3 = *reinterpret_cast<const uint4*>(W + (size_t)gb3 * K + k0 + bs3 * 8);
        return t;
    };
    auto store_tile = [&](int buf, const TileRegs& t) {
        *reinterpret_cast<uint4*>(&As[buf][ar0 * KSTR + as0 * 8]) = t.a0;
        *reinterpret_cast<uint4*>(&As[buf][ar1 * KSTR + as1 * 8]) = t.a1;
        *reinterpret_cast<uint4*>(&Bs[buf][br0 * KSTR + bs0 * 8]) = t.b0;
        *reinterpret_cast<uint4*>(&Bs[buf][br1 * KSTR + bs1 * 8]) = t.b1;
        *reinterpret_cast<uint4*>(&Bs[buf][br2 * KSTR + bs2 * 8]) = t.b2;
        *reinterpret_cast<uint4*>(&Bs[buf][br3 * KSTR + bs3 * 8]) = t.b3;
    };

    v8f acc[4];
    #pragma unroll
    for (int j = 0; j < 4; j++) acc[j] = zero8();

    {   // prologue: fill buffer 0
        TileRegs t = load_tile(0);
        store_tile(0, t);
    }
    __syncthreads();
    for (int k0 = 0; k0 < K; k0 += 64) {
        int buf = (k0 >> 6) & 1;
        bool more = (k0 + 64 < K);
        TileRegs nxt;
        if (more) nxt = load_tile(k0 + 64);  // global loads in flight during WMMAs below
        #pragma unroll
        for (int ks = 0; ks < 64; ks += 32) {
            union { v16bf v; uint4 q[2]; } af;
            const unsigned short* ap =
                &As[buf][(wm * 16 + (lane & 15)) * KSTR + ks + ((lane >= 16) ? 8 : 0)];
            af.q[0] = *reinterpret_cast<const uint4*>(ap);
            af.q[1] = *reinterpret_cast<const uint4*>(ap + 16);
            #pragma unroll
            for (int j = 0; j < 4; j++) {
                const unsigned short* bp =
                    &Bs[buf][(wn * 64 + j * 16 + (lane & 15)) * KSTR + ks +
                             ((lane >= 16) ? 8 : 0)];
                union { v16bf v; uint4 q[2]; } bfr;
                bfr.q[0] = *reinterpret_cast<const uint4*>(bp);
                bfr.q[1] = *reinterpret_cast<const uint4*>(bp + 16);
                acc[j] = __builtin_amdgcn_wmma_f32_16x16x32_bf16(false, af.v, false, bfr.v,
                                                                 (short)0, acc[j], false, false);
            }
        }
        if (more) store_tile(buf ^ 1, nxt);  // waits for loads here, after compute
        __syncthreads();
    }

    int colb  = lane & 15;
    int rbase = mbase + ((lane >= 16) ? 8 : 0);
    #pragma unroll
    for (int j = 0; j < 4; j++) {
        int col = nbase0 + j * 16 + colb;
        if (col >= N) continue;
        #pragma unroll
        for (int r = 0; r < 8; r++) {
            int rr = rbase + r;
            if (rr >= M) continue;
            size_t idx = (size_t)rr * N + col;
            float v = acc[j][r];
            if (mode == 0) {
                outF[idx] = v + (addC ? addC[idx] : 0.f);
            } else {
                float g = gateSrc[idx];
                float sg = g / (1.f + __expf(-g));
                outBF[idx] = f2bf(sg * v);
            }
        }
    }
}

// ---------------- packing (with RoPE) ----------------
// q: (T,1536) f32 -> qf: (B,H,S,96) bf16 with rotary on last 32 dims
__global__ __launch_bounds__(256) void pack_q_k(const float* q, unsigned short* qf,
                                                const int* startp) {
    int t = blockIdx.x; int b = t / SEQ, s = t - b * SEQ;
    float pos = (float)(startp[0] + s);
    for (int idx = threadIdx.x; idx < NH * 48; idx += 256) {
        int h = idx / 48, j = idx % 48;
        const float* qr = q + (size_t)t * (NH * QKD) + h * QKD;
        unsigned short* o = qf + (((size_t)b * NH + h) * SEQ + s) * QKD;
        if (j < 32) {
            o[2 * j]     = f2bf(qr[2 * j]);
            o[2 * j + 1] = f2bf(qr[2 * j + 1]);
        } else {
            int jr = j - 32;
            float inv = __powf(ROPE_BASE, -(2.f * jr) / (float)ROPE_D);
            float ang = pos * inv, c = __cosf(ang), sn = __sinf(ang);
            float x1 = qr[64 + 2 * jr], x2 = qr[64 + 2 * jr + 1];
            o[64 + 2 * jr]     = f2bf(x1 * c - x2 * sn);
            o[64 + 2 * jr + 1] = f2bf(x1 * sn + x2 * c);
        }
    }
}

// kva: (T,288) [256:288]=k_pe ; kvb: (T,2048) per head [0:64]=k_nope,[64:128]=v
// kp: (B,H,S,96) bf16 ; vt: (B,H,64,S) bf16 (transposed for contiguous-K B-frags)
__global__ __launch_bounds__(256) void pack_kv_k(const float* kva, const float* kvb,
                                                 unsigned short* kp, unsigned short* vt,
                                                 const int* startp) {
    int t = blockIdx.x; int b = t / SEQ, s = t - b * SEQ;
    __shared__ unsigned short pe[32];
    if (threadIdx.x < 16) {
        int jr = threadIdx.x;
        float pos = (float)(startp[0] + s);
        float inv = __powf(ROPE_BASE, -(2.f * jr) / (float)ROPE_D);
        float ang = pos * inv, c = __cosf(ang), sn = __sinf(ang);
        float x1 = kva[(size_t)t * 288 + 256 + 2 * jr];
        float x2 = kva[(size_t)t * 288 + 256 + 2 * jr + 1];
        pe[2 * jr]     = f2bf(x1 * c - x2 * sn);
        pe[2 * jr + 1] = f2bf(x1 * sn + x2 * c);
    }
    __syncthreads();
    for (int idx = threadIdx.x; idx < NH * 64; idx += 256) {
        int h = idx / 64, d = idx % 64;
        size_t base = (((size_t)b * NH + h) * SEQ + s) * QKD;
        kp[base + d] = f2bf(kvb[(size_t)t * 2048 + h * 128 + d]);
        vt[(((size_t)b * NH + h) * VD + d) * SEQ + s] =
            f2bf(kvb[(size_t)t * 2048 + h * 128 + 64 + d]);
    }
    for (int idx = threadIdx.x; idx < NH * 32; idx += 256) {
        int h = idx / 32, d = idx % 32;
        kp[(((size_t)b * NH + h) * SEQ + s) * QKD + 64 + d] = pe[d];
    }
}

// ---------------- flash-style MLA attention ----------------
// grid (SEQ/128, B*H), block 256; each wave owns 16 query rows, online softmax.
// Per-wave LDS scratch + s_wait_dscnt ordering (no block barrier: causal trip counts differ).
__global__ __launch_bounds__(256, 1) void attn_k(const unsigned short* __restrict__ qf,
                                                 const unsigned short* __restrict__ kp,
                                                 const unsigned short* __restrict__ vt,
                                                 unsigned short* __restrict__ ob) {
    __shared__ float          s_sc[8][16 * 32];
    __shared__ unsigned short s_p [8][16 * 32];
    __shared__ float          s_rs[8][16];
    int bh = blockIdx.y; int b = bh >> 4, h = bh & 15;
    int wave = threadIdx.x >> 5, lane = threadIdx.x & 31;
    int s0 = blockIdx.x * 128 + wave * 16;
    const unsigned short* Q  = qf + (size_t)bh * SEQ * QKD;
    const unsigned short* Kp = kp + (size_t)bh * SEQ * QKD;
    const unsigned short* Vt = vt + (size_t)bh * VD * SEQ;
    int mrow = s0 + (lane & 15);
    v16bf qa[3];
    #pragma unroll
    for (int kk = 0; kk < 3; kk++) qa[kk] = load_a_frag(Q, mrow, QKD, kk * 32, lane);
    v8f oacc[4];
    #pragma unroll
    for (int j = 0; j < 4; j++) oacc[j] = zero8();
    float mrun = -1e30f, lrun = 0.f;
    int myrow = lane & 15;
    int rl = (lane >= 16) ? 8 : 0;
    float* scw = s_sc[wave];
    unsigned short* pw = s_p[wave];
    const float scale = 0.1020620726f;   // 1/sqrt(96)
    int tend = s0 + 16;                  // causal bound (exclusive)
    for (int tb = 0; tb < tend; tb += 32) {
        // --- scores 16x32 via WMMA ---
        v8f sa[2]; sa[0] = zero8(); sa[1] = zero8();
        #pragma unroll
        for (int kk = 0; kk < 3; kk++) {
            #pragma unroll
            for (int c = 0; c < 2; c++) {
                v16bf bfr = load_b_frag(Kp, tb + c * 16 + (lane & 15), QKD, kk * 32, lane);
                sa[c] = __builtin_amdgcn_wmma_f32_16x16x32_bf16(false, qa[kk], false, bfr,
                                                                (short)0, sa[c], false, false);
            }
        }
        #pragma unroll
        for (int c = 0; c < 2; c++) {
            #pragma unroll
            for (int r = 0; r < 8; r++) {
                int lr = r + rl;
                int col = c * 16 + (lane & 15);
                float v = sa[c][r] * scale;
                if (tb + col > s0 + lr) v = -1e9f;
                scw[lr * 32 + col] = v;
            }
        }
        lds_fence();
        // --- per-row online softmax (row = lane&15, duplicated in both halves) ---
        float tm = -1e30f;
        for (int i = 0; i < 32; i++) tm = fmaxf(tm, scw[myrow * 32 + i]);
        float mnew = fmaxf(mrun, tm);
        float corr = __expf(mrun - mnew);
        float ls = 0.f;
        for (int i = 0; i < 32; i++) {
            float e = __expf(scw[myrow * 32 + i] - mnew);
            ls += e;
            pw[myrow * 32 + i] = f2bf(e);
        }
        lrun = lrun * corr + ls;
        mrun = mnew;
        s_rs[wave][myrow] = corr;
        lds_fence();
        // --- rescale O accumulators ---
        #pragma unroll
        for (int j = 0; j < 4; j++)
            #pragma unroll
            for (int r = 0; r < 8; r++)
                oacc[j][r] *= s_rs[wave][r + rl];
        // --- P @ V ---
        union { v16bf v; unsigned short u[16]; } af;
        {
            int lr2 = lane & 15;
            int koffb = (lane >= 16) ? 8 : 0;
            #pragma unroll
            for (int hh = 0; hh < 16; hh++) {
                int kk2 = (hh < 8 ? hh : hh + 8) + koffb;
                af.u[hh] = pw[lr2 * 32 + kk2];
            }
        }
        #pragma unroll
        for (int j = 0; j < 4; j++) {
            v16bf bfr = load_b_frag(Vt, j * 16 + (lane & 15), SEQ, tb, lane);
            oacc[j] = __builtin_amdgcn_wmma_f32_16x16x32_bf16(false, af.v, false, bfr,
                                                              (short)0, oacc[j], false, false);
        }
        lds_fence();
    }
    // --- normalize + store bf16 to o (T, H*VD) ---
    s_rs[wave][myrow] = lrun;
    lds_fence();
    #pragma unroll
    for (int j = 0; j < 4; j++) {
        int d = j * 16 + (lane & 15);
        #pragma unroll
        for (int r = 0; r < 8; r++) {
            int lr = r + rl;
            int s = s0 + lr;
            float v = oacc[j][r] / (s_rs[wave][lr] + 1e-20f);
            ob[((size_t)(b * SEQ + s)) * (NH * VD) + h * VD + d] = f2bf(v);
        }
    }
}

// ---------------- MoE ----------------
// 16 tokens x 16 experts per block; softmax + top2
__global__ __launch_bounds__(256) void moe_gate_k(const float* xf, const float* gate,
                                                  int* idx, float* wgt) {
    int tt = blockIdx.x * 16 + (threadIdx.x >> 4);
    int e  = threadIdx.x & 15;
    __shared__ float sc[16][16];
    const float* x = xf + (size_t)tt * DIMD;
    const float* g = gate + (size_t)e * DIMD;
    float s = 0.f;
    for (int i = 0; i < DIMD; i++) s += x[i] * g[i];
    sc[threadIdx.x >> 4][e] = s;
    __syncthreads();
    if (e == 0) {
        float* row = sc[threadIdx.x >> 4];
        float m = row[0];
        for (int i = 1; i < 16; i++) m = fmaxf(m, row[i]);
        float sum = 0.f;
        for (int i = 0; i < 16; i++) { row[i] = __expf(row[i] - m); sum += row[i]; }
        int i0 = 0; float v0 = -1.f;
        for (int i = 0; i < 16; i++) if (row[i] > v0) { v0 = row[i]; i0 = i; }
        int i1 = 0; float v1 = -1.f;
        for (int i = 0; i < 16; i++) if (i != i0 && row[i] > v1) { v1 = row[i]; i1 = i; }
        idx[tt * 2]     = i0;  idx[tt * 2 + 1] = i1;
        wgt[tt * 2]     = v0 / sum;
        wgt[tt * 2 + 1] = v1 / sum;
    }
}

// grid (T, 2): one block per (token, expert slot)
__global__ __launch_bounds__(256) void moe_expert_k(const float* xf,
                                                    const float* we1, const float* we2,
                                                    const float* we3,
                                                    const int* idx, const float* wgt, float* y) {
    int t = blockIdx.x, slot = blockIdx.y;
    int e = idx[t * 2 + slot];
    float cw = wgt[t * 2 + slot];
    const float* x = xf + (size_t)t * DIMD;
    __shared__ float sg[MOE_I];
    const float* W1 = we1 + (size_t)e * MOE_I * DIMD;
    const float* W3 = we3 + (size_t)e * MOE_I * DIMD;
    for (int f = threadIdx.x; f < MOE_I; f += 256) {
        const float* w1r = W1 + (size_t)f * DIMD;
        const float* w3r = W3 + (size_t)f * DIMD;
        float a = 0.f, bv = 0.f;
        for (int i = 0; i < DIMD; i++) { float xv = x[i]; a += xv * w1r[i]; bv += xv * w3r[i]; }
        float s = a / (1.f + __expf(-a));
        sg[f] = s * bv;
    }
    __syncthreads();
    const float* W2 = we2 + (size_t)e * DIMD * MOE_I;
    for (int d = threadIdx.x; d < DIMD; d += 256) {
        const float* w2r = W2 + (size_t)d * MOE_I;
        float a = 0.f;
        for (int i = 0; i < MOE_I; i++) a += sg[i] * w2r[i];
        atomicAdd(&y[(size_t)t * DIMD + d], cw * a);
    }
}

// ---------------- head ----------------
__global__ __launch_bounds__(256) void head_k(const float* hb, const float* normw,
                                              const float* hw, const float* hbias, float* out) {
    int b = blockIdx.x;
    const float* row = hb + ((size_t)(b * SEQ + SEQ - 1)) * DIMD;
    __shared__ float red[256];
    __shared__ float xr[DIMD];
    float s = 0.f;
    for (int i = threadIdx.x; i < DIMD; i += 256) { float v = row[i]; s += v * v; }
    red[threadIdx.x] = s; __syncthreads();
    for (int o = 128; o > 0; o >>= 1) {
        if (threadIdx.x < o) red[threadIdx.x] += red[threadIdx.x + o];
        __syncthreads();
    }
    float r = rsqrtf(red[0] / (float)DIMD + 1e-6f);
    for (int i = threadIdx.x; i < DIMD; i += 256) xr[i] = row[i] * r * normw[i];
    __syncthreads();
    for (int o = threadIdx.x; o < OUTD; o += 256) {
        const float* wr = hw + (size_t)o * DIMD;
        float a = 0.f;
        for (int i = 0; i < DIMD; i++) a += xr[i] * wr[i];
        out[(size_t)b * OUTD + o] = a + hbias[o];
    }
}

// ---------------- host orchestration ----------------
extern "C" void kernel_launch(void* const* d_in, const int* in_sizes, int n_in,
                              void* d_out, int out_size, void* d_ws, size_t ws_size,
                              hipStream_t stream) {
    (void)in_sizes; (void)n_in; (void)out_size; (void)ws_size;
    const float* h_in   = (const float*)d_in[0];
    const float* Lwq[2]     = {(const float*)d_in[1],  (const float*)d_in[11]};
    const float* Lwkva[2]   = {(const float*)d_in[2],  (const float*)d_in[12]};
    const float* Lkvnorm[2] = {(const float*)d_in[3],  (const float*)d_in[13]};
    const float* Lwkvb[2]   = {(const float*)d_in[4],  (const float*)d_in[14]};
    const float* Lwo[2]     = {(const float*)d_in[5],  (const float*)d_in[15]};
    const float* Lanorm[2]  = {(const float*)d_in[6],  (const float*)d_in[16]};
    const float* Lfnorm[2]  = {(const float*)d_in[7],  (const float*)d_in[17]};
    const float* w1   = (const float*)d_in[8];
    const float* w2   = (const float*)d_in[9];
    const float* w3   = (const float*)d_in[10];
    const float* gate = (const float*)d_in[18];
    const float* we1  = (const float*)d_in[19];
    const float* we2  = (const float*)d_in[20];
    const float* we3  = (const float*)d_in[21];
    const float* ws1  = (const float*)d_in[22];
    const float* ws2  = (const float*)d_in[23];
    const float* ws3  = (const float*)d_in[24];
    const float* fnorm = (const float*)d_in[25];
    const float* headw = (const float*)d_in[26];
    const float* headb = (const float*)d_in[27];
    const int*   startp = (const int*)d_in[28];
    float* outp = (float*)d_out;

    // workspace bump allocator
    char* ws = (char*)d_ws;
    size_t off = 0;
    auto alloc = [&](size_t bytes) -> char* {
        char* p = ws + off;
        off += (bytes + 255) & ~(size_t)255;
        return p;
    };
    // persistent
    float*          hbuf = (float*)         alloc((size_t)TTOK * DIMD * 4);
    unsigned short* xnb  = (unsigned short*)alloc((size_t)TTOK * DIMD * 2);
    float*          xf   = (float*)         alloc((size_t)TTOK * DIMD * 4);
    // bf16 weight arena (reused per layer)
    unsigned short* wqb  = (unsigned short*)alloc((size_t)NH * QKD * DIMD * 2);
    unsigned short* wkab = (unsigned short*)alloc((size_t)(KVL + ROPE_D) * DIMD * 2);
    unsigned short* wkbb = (unsigned short*)alloc((size_t)NH * 128 * KVL * 2);
    unsigned short* wob  = (unsigned short*)alloc((size_t)DIMD * DIMD * 2);
    unsigned short* wf1  = (unsigned short*)alloc((size_t)INTER_D * DIMD * 2);
    unsigned short* wf2  = (unsigned short*)alloc((size_t)INTER_D * DIMD * 2);
    unsigned short* wf3  = (unsigned short*)alloc((size_t)INTER_D * DIMD * 2);
    // phase scratch (A and B aliased from same base)
    size_t pbase = off;
    // phase A
    float*          qb    = (float*)         (ws + pbase);
    float*          kva   = (float*)         ((char*)qb    + (size_t)TTOK * (NH*QKD) * 4);
    unsigned short* cb    = (unsigned short*)((char*)kva   + (size_t)TTOK * 288 * 4);
    float*          kvbuf = (float*)         ((char*)cb    + (size_t)TTOK * KVL * 2);
    unsigned short* qfb   = (unsigned short*)((char*)kvbuf + (size_t)TTOK * 2048 * 4);
    unsigned short* kpb   = (unsigned short*)((char*)qfb   + (size_t)BATCH*NH*SEQ*QKD * 2);
    unsigned short* vtb   = (unsigned short*)((char*)kpb   + (size_t)BATCH*NH*SEQ*QKD * 2);
    unsigned short* obuf  = (unsigned short*)((char*)vtb   + (size_t)BATCH*NH*VD*SEQ * 2);
    // phase B (overlaps phase A)
    float*          g1   = (float*)         (ws + pbase);
    unsigned short* ub   = (unsigned short*)((char*)g1  + (size_t)TTOK * INTER_D * 4);
    float*          gs1  = (float*)         ((char*)ub  + (size_t)TTOK * INTER_D * 2);
    unsigned short* usb  = (unsigned short*)((char*)gs1 + (size_t)TTOK * MOE_I * 4);
    float*          yb   = (float*)         ((char*)usb + (size_t)TTOK * MOE_I * 2);
    int*            idxb = (int*)            ((char*)yb  + (size_t)TTOK * DIMD * 4);
    float*          wgtb = (float*)          ((char*)idxb + (size_t)TTOK * 2 * 4);

    auto conv = [&](const float* s, unsigned short* d, size_t n) {
        f32_to_bf16_k<<<(unsigned)((n + 255) / 256), 256, 0, stream>>>(s, d, n);
    };
    auto gemm = [&](const unsigned short* A, const unsigned short* W,
                    int M, int N, int K, const float* addC, const float* gateS,
                    float* oF, unsigned short* oB, int mode) {
        dim3 g((N + 127) / 128, (M + 63) / 64);
        gemm_bf16_k<<<g, 256, 0, stream>>>(A, W, M, N, K, addC, gateS, oF, oB, mode);
    };

    // residual init
    hipMemcpyAsync(hbuf, h_in, (size_t)TTOK * DIMD * 4, hipMemcpyDeviceToDevice, stream);

    for (int lid = 0; lid < 2; lid++) {
        // convert attention weights
        conv(Lwq[lid],   wqb,  (size_t)NH * QKD * DIMD);
        conv(Lwkva[lid], wkab, (size_t)(KVL + ROPE_D) * DIMD);
        conv(Lwkvb[lid], wkbb, (size_t)NH * 128 * KVL);
        conv(Lwo[lid],   wob,  (size_t)DIMD * DIMD);
        // attention
        rmsnorm_k<<<TTOK, 256, 0, stream>>>(hbuf, Lanorm[lid], DIMD, DIMD, xnb, nullptr);
        gemm(xnb, wqb,  TTOK, NH * QKD,     DIMD, nullptr, nullptr, qb,  nullptr, 0);
        gemm(xnb, wkab, TTOK, KVL + ROPE_D, DIMD, nullptr, nullptr, kva, nullptr, 0);
        rmsnorm_k<<<TTOK, 256, 0, stream>>>(kva, Lkvnorm[lid], KVL, 288, cb, nullptr);
        gemm(cb, wkbb, TTOK, NH * 128, KVL, nullptr, nullptr, kvbuf, nullptr, 0);
        pack_q_k<<<TTOK, 256, 0, stream>>>(qb, qfb, startp);
        pack_kv_k<<<TTOK, 256, 0, stream>>>(kva, kvbuf, kpb, vtb, startp);
        attn_k<<<dim3(SEQ / 128, BATCH * NH), 256, 0, stream>>>(qfb, kpb, vtb, obuf);
        gemm(obuf, wob, TTOK, DIMD, NH * VD, hbuf, nullptr, hbuf, nullptr, 0);
        // ffn / moe
        if (lid == 0) {
            rmsnorm_k<<<TTOK, 256, 0, stream>>>(hbuf, Lfnorm[lid], DIMD, DIMD, xnb, nullptr);
            conv(w1, wf1, (size_t)INTER_D * DIMD);
            conv(w3, wf3, (size_t)INTER_D * DIMD);
            conv(w2, wf2, (size_t)INTER_D * DIMD);
            gemm(xnb, wf1, TTOK, INTER_D, DIMD, nullptr, nullptr, g1, nullptr, 0);
            gemm(xnb, wf3, TTOK, INTER_D, DIMD, nullptr, g1, nullptr, ub, 2);
            gemm(ub,  wf2, TTOK, DIMD, INTER_D, hbuf, nullptr, hbuf, nullptr, 0);
        } else {
            rmsnorm_k<<<TTOK, 256, 0, stream>>>(hbuf, Lfnorm[lid], DIMD, DIMD, xnb, xf);
            moe_gate_k<<<TTOK / 16, 256, 0, stream>>>(xf, gate, idxb, wgtb);
            zero_f32_k<<<(TTOK * DIMD + 255) / 256, 256, 0, stream>>>(yb, (size_t)TTOK * DIMD);
            moe_expert_k<<<dim3(TTOK, 2), 256, 0, stream>>>(xf, we1, we2, we3, idxb, wgtb, yb);
            conv(ws1, wf1, (size_t)MOE_I * DIMD);
            conv(ws3, wf3, (size_t)MOE_I * DIMD);
            conv(ws2, wf2, (size_t)MOE_I * DIMD);
            gemm(xnb, wf1, TTOK, MOE_I, DIMD, nullptr, nullptr, gs1, nullptr, 0);
            gemm(xnb, wf3, TTOK, MOE_I, DIMD, nullptr, gs1, nullptr, usb, 2);
            gemm(usb, wf2, TTOK, DIMD, MOE_I, yb, nullptr, yb, nullptr, 0);
            add_f32_k<<<(TTOK * DIMD + 255) / 256, 256, 0, stream>>>(hbuf, yb,
                                                                     (size_t)TTOK * DIMD);
        }
    }
    head_k<<<BATCH, 256, 0, stream>>>(hbuf, fnorm, headw, headb, outp);
}
